// Head_85736137163193
// MI455X (gfx1250) — compile-verified, matrix operand order checked by reading, not run on hardware
//
#include <hip/hip_runtime.h>
#include <hip/hip_bf16.h>
#include <math.h>

// Shapes (compile-time constants from the reference)
#define HEAD_SIZE 64
#define N_EMBED   1024
#define BLOCK_T   2048
#define BATCH     8
#define BT        (BATCH * BLOCK_T)   // 16384 total rows

typedef __attribute__((ext_vector_type(16))) __bf16 v16bf;
typedef __attribute__((ext_vector_type(8)))  float  v8f;
typedef unsigned int u32x4 __attribute__((ext_vector_type(4)));
typedef int          i32x8 __attribute__((ext_vector_type(8)));
typedef int          i32x4 __attribute__((ext_vector_type(4)));

__device__ __forceinline__ v8f wmma_bf16(v16bf a, v16bf b, v8f c) {
    // D = A(16x32 bf16) * B(32x16 bf16) + C(16x16 f32)
    return __builtin_amdgcn_wmma_f32_16x16x32_bf16(
        false, a, false, b, (short)0, c, false, false);
}

// A-matrix (16x32 bf16) fragment from row-major LDS storage.
// Per ISA 7.12.2: lane L holds row M=L%16; lanes 16-31 K-offset by 8;
// VGPR v holds K pair (v<4 ? 2v : 16+2(v-4)).
// Also used for B fragments when B's column n is stored as contiguous row n.
__device__ __forceinline__ v16bf frag_rows(const __bf16* base, int ld, int row0, int k0) {
    const int lane = threadIdx.x & 31;
    const int g = lane >> 4;
    const __bf16* p = base + (size_t)(row0 + (lane & 15)) * ld + k0 + 8 * g;
    v16bf f;
#pragma unroll
    for (int v = 0; v < 8; ++v) {
        const int kk = (v < 4) ? (2 * v) : (16 + 2 * (v - 4));
        f[2 * v]     = p[kk];
        f[2 * v + 1] = p[kk + 1];
    }
    return f;
}

// B-matrix (32x16 bf16) fragment where B is stored row-major as [K][N].
__device__ __forceinline__ v16bf frag_cols(const __bf16* base, int ld, int col0, int k0) {
    const int lane = threadIdx.x & 31;
    const int g = lane >> 4;
    const __bf16* p = base + (size_t)(k0 + 8 * g) * ld + col0 + (lane & 15);
    v16bf f;
#pragma unroll
    for (int v = 0; v < 8; ++v) {
        const int kk = (v < 4) ? (2 * v) : (16 + 2 * (v - 4));
        f[2 * v]     = p[(size_t)kk * ld];
        f[2 * v + 1] = p[(size_t)(kk + 1) * ld];
    }
    return f;
}

// ---------------------------------------------------------------------------
// Tensor Data Mover: DMA a 64x64 bf16 tile (row stride 64 elem in memory) into
// LDS, padding 4 DWORDs (8 bf16) after every 32 DWORDs (one 64-elem row) so
// the LDS image has row stride 72 bf16 — matching the padded compute layout.
// Descriptor per CDNA5 ISA 8.3-8.6 (D# groups). Issued by one wave.
// ---------------------------------------------------------------------------
__device__ __forceinline__ void tdm_load_tile_64x64(unsigned lds_off, const __bf16* gptr) {
    const unsigned long long ga = (unsigned long long)(uintptr_t)gptr;
    u32x4 g0;
    g0[0] = 1u;                                      // count=1, user descriptor
    g0[1] = lds_off;                                 // lds_addr (bytes)
    g0[2] = (unsigned)(ga & 0xffffffffu);            // global_addr[31:0]
    g0[3] = (unsigned)((ga >> 32) & 0x01ffffffu)     // global_addr[56:32]
          | (2u << 30);                              // type=2 ("image")
    i32x8 g1;
    g1[0] = (int)((1u << 16)      // data_size = 1 -> 2 bytes/elem
                | (1u << 20)      // pad_enable
                | (4u << 22)      // pad_interval = 4 -> every 32 DWORDs
                | (3u << 25));    // pad_amount  = 3 -> 4 DWORDs (8 bf16)
    g1[1] = (int)(64u << 16);     // tensor_dim0 low16 = 64 (bits 79:48)
    g1[2] = (int)(64u << 16);     // tensor_dim0 hi = 0; tensor_dim1 low16 = 64
    g1[3] = (int)(64u << 16);     // tensor_dim1 hi = 0; tile_dim0 = 64
    g1[4] = 64;                   // tile_dim1 = 64; tile_dim2 = 0
    g1[5] = 64;                   // tensor_dim0_stride low32 = 64 elems
    g1[6] = 0;                    // stride hi; tensor_dim1_stride low16 = 0
    g1[7] = 0;
    i32x4 z4 = {};
    i32x8 z8 = {};
    // 6-arg form (this toolchain): (g0, g1, g2, g3, g4, cpol)
    __builtin_amdgcn_tensor_load_to_lds(g0, g1, z4, z4, z8, 0);
}

// ---------------------------------------------------------------------------
// Kernel 1: QKV projection.  out[row, h] = sum_c X[row, c] * W[h, c]
// Block: 256 threads = 8 waves arranged 2(M) x 4(N); tile 32x64, K-step 64
// (2 WMMAs per barrier pair). fp32 -> bf16 conversion during LDS staging.
// ---------------------------------------------------------------------------
__global__ __launch_bounds__(256) void qkv_proj_kernel(
    const float* __restrict__ xin, const float* __restrict__ xmem,
    const float* __restrict__ Wq, const float* __restrict__ Wk,
    const float* __restrict__ Wv,
    __bf16* __restrict__ Qo, __bf16* __restrict__ Ko, __bf16* __restrict__ Vo) {

    __shared__ __align__(16) __bf16 As[32][72];   // activation tile (bf16)
    __shared__ __align__(16) __bf16 Wsh[64][72];  // weight tile, row h contiguous in K

    const int proj = blockIdx.y;
    const float* X = (proj == 0) ? xin : xmem;
    const float* W = (proj == 0) ? Wq : ((proj == 1) ? Wk : Wv);
    __bf16* O      = (proj == 0) ? Qo : ((proj == 1) ? Ko : Vo);

    const int m0   = blockIdx.x * 32;
    const int tid  = threadIdx.x;
    const int wave = tid >> 5;
    const int lane = tid & 31;
    const int wm   = wave >> 2;      // 0..1 (M tile)
    const int wn   = wave & 3;       // 0..3 (N tile)

    v8f acc = {};

    for (int k0 = 0; k0 < N_EMBED; k0 += 64) {
        __syncthreads();
#pragma unroll
        for (int j = 0; j < 2; ++j) {   // A: 32 rows x 64 cols = 512 float4
            const int idx = tid + j * 256;
            const int row = idx >> 4;
            const int c4  = (idx & 15) * 4;
            float4 f = *(const float4*)(X + (size_t)(m0 + row) * N_EMBED + k0 + c4);
            As[row][c4 + 0] = (__bf16)f.x; As[row][c4 + 1] = (__bf16)f.y;
            As[row][c4 + 2] = (__bf16)f.z; As[row][c4 + 3] = (__bf16)f.w;
        }
#pragma unroll
        for (int j = 0; j < 4; ++j) {   // W: 64 rows x 64 cols = 1024 float4
            const int idx = tid + j * 256;
            const int row = idx >> 4;
            const int c4  = (idx & 15) * 4;
            float4 f = *(const float4*)(W + (size_t)row * N_EMBED + k0 + c4);
            Wsh[row][c4 + 0] = (__bf16)f.x; Wsh[row][c4 + 1] = (__bf16)f.y;
            Wsh[row][c4 + 2] = (__bf16)f.z; Wsh[row][c4 + 3] = (__bf16)f.w;
        }
        __syncthreads();

        v16bf a0 = frag_rows(&As[0][0],  72, wm * 16, 0);
        v16bf b0 = frag_rows(&Wsh[0][0], 72, wn * 16, 0);
        acc = wmma_bf16(a0, b0, acc);
        v16bf a1 = frag_rows(&As[0][0],  72, wm * 16, 32);
        v16bf b1 = frag_rows(&Wsh[0][0], 72, wn * 16, 32);
        acc = wmma_bf16(a1, b1, acc);
    }

    const int g = lane >> 4;
#pragma unroll
    for (int r = 0; r < 8; ++r) {
        const int row = m0 + wm * 16 + r + 8 * g;
        const int col = wn * 16 + (lane & 15);
        O[(size_t)row * HEAD_SIZE + col] = (__bf16)acc[r];
    }
}

// ---------------------------------------------------------------------------
// Kernel 2: causal flash attention over bf16 Q/K/V, fp32 output.
// Block: 128 threads = 4 waves; block owns (batch b, 64 query rows).
// Q/K/V tiles staged LDS via Tensor Data Mover (wave 0 issues, TENSORcnt wait,
// workgroup barrier releases). Each wave owns 16 query rows x full head.
// ---------------------------------------------------------------------------
__global__ __launch_bounds__(128) void attn_kernel(
    const __bf16* __restrict__ Q, const __bf16* __restrict__ K,
    const __bf16* __restrict__ V, float* __restrict__ out) {

    __shared__ __align__(16) __bf16 Qs[64][72];
    __shared__ __align__(16) __bf16 Ks[64][72];
    __shared__ __align__(16) __bf16 Vs[64][72];
    __shared__ __align__(16) __bf16 Ps[4][16][72];  // per-wave P staging

    const int b    = blockIdx.y;
    const int q0   = blockIdx.x * 64;
    const int tid  = threadIdx.x;
    const int wave = tid >> 5;
    const int lane = tid & 31;
    const int g    = lane >> 4;
    const int ln   = lane & 15;

    const unsigned ldsQ = (unsigned)(uintptr_t)&Qs[0][0];
    const unsigned ldsK = (unsigned)(uintptr_t)&Ks[0][0];
    const unsigned ldsV = (unsigned)(uintptr_t)&Vs[0][0];

    const size_t baseQ = ((size_t)b * BLOCK_T + q0) * HEAD_SIZE;

    // stage Q (64x64 bf16) via TDM
    if (wave == 0) {
        tdm_load_tile_64x64(ldsQ, Q + baseQ);
        __builtin_amdgcn_s_wait_tensorcnt(0);
    }
    __syncthreads();

    const v16bf qa0 = frag_rows(&Qs[0][0], 72, wave * 16, 0);
    const v16bf qa1 = frag_rows(&Qs[0][0], 72, wave * 16, 32);

    v8f o[4] = {};
    float mrow[8], lrow[8];
#pragma unroll
    for (int r = 0; r < 8; ++r) { mrow[r] = -3.0e38f; lrow[r] = 0.0f; }

    const float scale = 0.125f;  // 1/sqrt(64)

    for (int s0 = 0; s0 <= q0; s0 += 64) {
        __syncthreads();
        const size_t baseK = ((size_t)b * BLOCK_T + s0) * HEAD_SIZE;
        if (wave == 0) {
            tdm_load_tile_64x64(ldsK, K + baseK);
            tdm_load_tile_64x64(ldsV, V + baseK);
            __builtin_amdgcn_s_wait_tensorcnt(0);
        }
        __syncthreads();

        const bool need_mask = (s0 + 64 > q0);  // only the diagonal block

        float p[4][8];
        float tmax[8];
#pragma unroll
        for (int r = 0; r < 8; ++r) tmax[r] = -3.0e38f;

        // scores S = Q * K^T  (B col n = K row s0+n, contiguous in LDS)
#pragma unroll
        for (int nt = 0; nt < 4; ++nt) {
            v8f s = {};
            v16bf kb0 = frag_rows(&Ks[0][0], 72, nt * 16, 0);
            v16bf kb1 = frag_rows(&Ks[0][0], 72, nt * 16, 32);
            s = wmma_bf16(qa0, kb0, s);
            s = wmma_bf16(qa1, kb1, s);
#pragma unroll
            for (int r = 0; r < 8; ++r) {
                float val = s[r] * scale;
                if (need_mask) {
                    const int qrow = q0 + wave * 16 + r + 8 * g;
                    const int krow = s0 + nt * 16 + ln;
                    if (krow > qrow) val = -3.0e38f;
                }
                p[nt][r] = val;
                tmax[r]  = fmaxf(tmax[r], val);
            }
        }

        // row max across the 16 lanes holding one row (half-wave reduction)
#pragma unroll
        for (int off = 1; off < 16; off <<= 1) {
#pragma unroll
            for (int r = 0; r < 8; ++r)
                tmax[r] = fmaxf(tmax[r], __shfl_xor(tmax[r], off, 32));
        }

        float fac[8], rs[8];
#pragma unroll
        for (int r = 0; r < 8; ++r) {
            const float nm = fmaxf(mrow[r], tmax[r]);
            fac[r] = __expf(mrow[r] - nm);
            mrow[r] = nm;
            float rsum = 0.0f;
#pragma unroll
            for (int nt = 0; nt < 4; ++nt) {
                const float e = __expf(p[nt][r] - nm);
                p[nt][r] = e;
                rsum += e;
            }
            rs[r] = rsum;
        }
#pragma unroll
        for (int off = 1; off < 16; off <<= 1) {
#pragma unroll
            for (int r = 0; r < 8; ++r)
                rs[r] += __shfl_xor(rs[r], off, 32);
        }
#pragma unroll
        for (int r = 0; r < 8; ++r) {
            lrow[r] = lrow[r] * fac[r] + rs[r];
#pragma unroll
            for (int nt = 0; nt < 4; ++nt) o[nt][r] *= fac[r];
        }

        // round-trip P through per-wave LDS: C layout -> A layout
#pragma unroll
        for (int nt = 0; nt < 4; ++nt)
#pragma unroll
            for (int r = 0; r < 8; ++r)
                Ps[wave][r + 8 * g][nt * 16 + ln] = (__bf16)p[nt][r];
        asm volatile("s_wait_dscnt 0" ::: "memory");

        const v16bf pa0 = frag_rows(&Ps[wave][0][0], 72, 0, 0);
        const v16bf pa1 = frag_rows(&Ps[wave][0][0], 72, 0, 32);
#pragma unroll
        for (int nt = 0; nt < 4; ++nt) {
            v16bf vb0 = frag_cols(&Vs[0][0], 72, nt * 16, 0);
            v16bf vb1 = frag_cols(&Vs[0][0], 72, nt * 16, 32);
            o[nt] = wmma_bf16(pa0, vb0, o[nt]);
            o[nt] = wmma_bf16(pa1, vb1, o[nt]);
        }
    }

    // epilogue: out[b, t, h] = o / rowsum   (fp32)
#pragma unroll
    for (int nt = 0; nt < 4; ++nt)
#pragma unroll
        for (int r = 0; r < 8; ++r) {
            const int t = q0 + wave * 16 + r + 8 * g;
            const int h = nt * 16 + ln;
            out[((size_t)b * BLOCK_T + t) * HEAD_SIZE + h] = o[nt][r] / lrow[r];
        }
}

// ---------------------------------------------------------------------------
extern "C" void kernel_launch(void* const* d_in, const int* in_sizes, int n_in,
                              void* d_out, int out_size, void* d_ws, size_t ws_size,
                              hipStream_t stream) {
    const float* index  = (const float*)d_in[0];  // [B, T, C]
    const float* memory = (const float*)d_in[1];  // [B, S, C]
    const float* Wq     = (const float*)d_in[2];  // [H, C]
    const float* Wk     = (const float*)d_in[3];
    const float* Wv     = (const float*)d_in[4];
    float* out          = (float*)d_out;          // [B, T, H] fp32

    // workspace: bf16 Q, K, V  (3 * 16384 * 64 * 2B = 6 MB)
    __bf16* Qw = (__bf16*)d_ws;
    __bf16* Kw = Qw + (size_t)BT * HEAD_SIZE;
    __bf16* Vw = Kw + (size_t)BT * HEAD_SIZE;

    qkv_proj_kernel<<<dim3(BT / 32, 3), 256, 0, stream>>>(
        index, memory, Wq, Wk, Wv, Qw, Kw, Vw);

    attn_kernel<<<dim3(BLOCK_T / 64, BATCH), 128, 0, stream>>>(Qw, Kw, Vw, out);
}
